// CAttention_88218628260008
// MI455X (gfx1250) — matrix operand order repaired and checked by reading, not compiled, not run on hardware
//
#include <hip/hip_runtime.h>
#include <hip/hip_bf16.h>

#define B_ 4
#define S_ 2048
#define D_ 1024

typedef __attribute__((ext_vector_type(16))) __bf16 v16bf;
typedef __attribute__((ext_vector_type(8)))  __bf16 v8bf;
typedef __attribute__((ext_vector_type(4)))  __bf16 v4bf;
typedef __attribute__((ext_vector_type(8)))  float  v8f;

// ---------- helpers ----------

static __device__ __forceinline__ __bf16 f2bf(float f) {
  unsigned u = __float_as_uint(f);
  u += 0x7FFFu + ((u >> 16) & 1u);          // round-to-nearest-even
  unsigned short h = (unsigned short)(u >> 16);
  __bf16 r;
  __builtin_memcpy(&r, &h, sizeof(r));
  return r;
}

// A-operand fragment: 16x32 (MxK) bf16, memory row-major [M][K], leading dim ld.
// ISA layout: lane(0-15)=M, elems 0-7 -> K=half*8.., elems 8-15 -> K=16+half*8..
static __device__ __forceinline__ v16bf load_a_frag(const __bf16* base, int ld, int lane) {
  const int row = lane & 15;
  const int hf  = lane >> 4;
  const __bf16* p = base + (size_t)row * ld + hf * 8;
  v8bf lo = *(const v8bf*)(p);        // K = hf*8 .. hf*8+7
  v8bf hi = *(const v8bf*)(p + 16);   // K = 16+hf*8 .. 16+hf*8+7
  return __builtin_shufflevector(lo, hi, 0,1,2,3,4,5,6,7,8,9,10,11,12,13,14,15);
}

// B-operand fragment: 32x16 (KxN) bf16, memory holds B^T row-major [N][K], leading dim ld.
// ISA layout: lane(0-15)=N, elems 0-7 -> K=half*8.., elems 8-15 -> K=16+half*8..
static __device__ __forceinline__ v16bf load_b_frag(const __bf16* base, int ld, int lane) {
  const int row = lane & 15;
  const int hf  = lane >> 4;
  const __bf16* p = base + (size_t)row * ld + hf * 8;
  v8bf lo = *(const v8bf*)(p);
  v8bf hi = *(const v8bf*)(p + 16);
  return __builtin_shufflevector(lo, hi, 0,1,2,3,4,5,6,7,8,9,10,11,12,13,14,15);
}

static __device__ __forceinline__ v8f wmma_bf16(v16bf a, v16bf b, v8f c) {
  return __builtin_amdgcn_wmma_f32_16x16x32_bf16(
      /*neg_a=*/false, a, /*neg_b=*/false, b,
      /*c_mod=*/(short)0, c, /*reuse_a=*/false, /*reuse_b=*/false);
}

// Register-blocked K-loop: BMxBN 16x16 tiles, A row-major, B given as B^T row-major.
template <int BM, int BN>
static __device__ __forceinline__ void gemm_tile(const __bf16* __restrict__ Abase, int lda,
                                                 const __bf16* __restrict__ Bbase, int ldb,
                                                 int K, int lane, v8f (&acc)[BM][BN]) {
#pragma unroll 2
  for (int kk = 0; kk < K; kk += 32) {
    v16bf a[BM], b[BN];
#pragma unroll
    for (int m = 0; m < BM; ++m)
      a[m] = load_a_frag(Abase + (size_t)(m * 16) * lda + kk, lda, lane);
#pragma unroll
    for (int n = 0; n < BN; ++n)
      b[n] = load_b_frag(Bbase + (size_t)(n * 16) * ldb + kk, ldb, lane);
#pragma unroll
    for (int m = 0; m < BM; ++m)
#pragma unroll
      for (int n = 0; n < BN; ++n)
        acc[m][n] = wmma_bf16(a[m], b[n], acc[m][n]);
  }
}

// ---------- kernel 1: x fp32 -> bf16 ----------

__global__ __launch_bounds__(256) void k_convert_x(const float* __restrict__ x,
                                                   __bf16* __restrict__ xb) {
  size_t i = ((size_t)blockIdx.x * 256 + threadIdx.x) * 4;
  float4 v = *(const float4*)(x + i);
  v4bf o;
  o[0] = f2bf(v.x); o[1] = f2bf(v.y); o[2] = f2bf(v.z); o[3] = f2bf(v.w);
  *(v4bf*)(xb + i) = o;
}

// ---------- kernel 2: W (D,D) fp32 -> W^T bf16, LDS-tiled ----------

__global__ __launch_bounds__(256) void k_transpose_w(const float* __restrict__ Wq,
                                                     const float* __restrict__ Wk,
                                                     const float* __restrict__ Wv,
                                                     __bf16* __restrict__ Wt) {
  __shared__ float t[16][17];
  const int tilesPerW = (D_ / 16) * (D_ / 16);
  int w  = blockIdx.x / tilesPerW;
  int r  = blockIdx.x % tilesPerW;
  int et = r & (D_ / 16 - 1);
  int dt = r >> 6;                       // D_/16 == 64
  int tx = threadIdx.x & 15, ty = threadIdx.x >> 4;
  const float* W = (w == 0) ? Wq : ((w == 1) ? Wk : Wv);
  t[ty][tx] = W[(size_t)(dt * 16 + ty) * D_ + et * 16 + tx];
  __syncthreads();
  Wt[(size_t)w * D_ * D_ + (size_t)(et * 16 + ty) * D_ + dt * 16 + tx] = f2bf(t[tx][ty]);
}

// ---------- kernel 3: QKV projection GEMM (one wave = 32x32 tile, 2x2 blocked) ----------
// Q: bf16 row-major [B*S][D], pre-scaled by 1/sqrt(D).  K: bf16 row-major.
// V: stored transposed VT[b][d][s] bf16 (contiguous v8bf store per lane).

__global__ __launch_bounds__(256) void k_qkv(const __bf16* __restrict__ xb,
                                             const __bf16* __restrict__ Wt,
                                             const float* __restrict__ bq,
                                             const float* __restrict__ bk,
                                             const float* __restrict__ bv,
                                             __bf16* __restrict__ Qb,
                                             __bf16* __restrict__ Kb,
                                             __bf16* __restrict__ VT) {
  const int MT = (B_ * S_) / 32;   // 256
  const int NT = D_ / 32;          // 32
  int wave = blockIdx.x * (blockDim.x >> 5) + (threadIdx.x >> 5);
  int lane = threadIdx.x & 31;
  int w  = wave / (MT * NT);
  int r  = wave % (MT * NT);
  int mt = r / NT, nt = r % NT;
  int gm = mt * 32, gn = nt * 32;

  const __bf16* A0 = xb + (size_t)gm * D_;
  const __bf16* B0 = Wt + (size_t)w * D_ * D_ + (size_t)gn * D_;

  v8f acc[2][2] = {};
  gemm_tile<2, 2>(A0, D_, B0, D_, D_, lane, acc);

  int n = lane & 15, hf = lane >> 4;
  const float* bias = (w == 0) ? bq : ((w == 1) ? bk : bv);

  if (w == 0) {
    const float scale = 0.03125f;        // 1/sqrt(1024)
#pragma unroll
    for (int mi = 0; mi < 2; ++mi)
#pragma unroll
      for (int ni = 0; ni < 2; ++ni) {
        int e = gn + ni * 16 + n;
        float bb = bias[e];
#pragma unroll
        for (int i = 0; i < 8; ++i) {
          int m = gm + mi * 16 + hf * 8 + i;
          Qb[(size_t)m * D_ + e] = f2bf((acc[mi][ni][i] + bb) * scale);
        }
      }
  } else if (w == 1) {
#pragma unroll
    for (int mi = 0; mi < 2; ++mi)
#pragma unroll
      for (int ni = 0; ni < 2; ++ni) {
        int e = gn + ni * 16 + n;
        float bb = bias[e];
#pragma unroll
        for (int i = 0; i < 8; ++i) {
          int m = gm + mi * 16 + hf * 8 + i;
          Kb[(size_t)m * D_ + e] = f2bf(acc[mi][ni][i] + bb);
        }
      }
  } else {
    int b  = gm / S_;
    int s0 = gm % S_;
#pragma unroll
    for (int mi = 0; mi < 2; ++mi)
#pragma unroll
      for (int ni = 0; ni < 2; ++ni) {
        int e = gn + ni * 16 + n;
        float bb = bias[e];
        v8bf o;
#pragma unroll
        for (int i = 0; i < 8; ++i) o[i] = f2bf(acc[mi][ni][i] + bb);
        *(v8bf*)(VT + (size_t)b * D_ * S_ + (size_t)e * S_ + s0 + mi * 16 + hf * 8) = o;
      }
  }
}

// ---------- kernel 4: scores = Q K^T (scale already in Q), 4x4 blocked (64x64/wave) ----

__global__ __launch_bounds__(256) void k_scores(const __bf16* __restrict__ Qb,
                                                const __bf16* __restrict__ Kb,
                                                float* __restrict__ Sc) {
  const int QT = S_ / 64;          // 32
  int wave = blockIdx.x * (blockDim.x >> 5) + (threadIdx.x >> 5);
  int lane = threadIdx.x & 31;
  int b  = wave / (QT * QT);
  int r  = wave % (QT * QT);
  int qt = r / QT, kt = r % QT;

  const __bf16* A0 = Qb + ((size_t)b * S_ + qt * 64) * D_;
  const __bf16* B0 = Kb + ((size_t)b * S_ + kt * 64) * D_;   // row-major K == B^T

  v8f acc[4][4] = {};
  gemm_tile<4, 4>(A0, D_, B0, D_, D_, lane, acc);

  int n = lane & 15, hf = lane >> 4;
  float* out = Sc + ((size_t)b * S_ + qt * 64) * S_ + kt * 64;
#pragma unroll
  for (int mi = 0; mi < 4; ++mi)
#pragma unroll
    for (int ni = 0; ni < 4; ++ni)
#pragma unroll
      for (int i = 0; i < 8; ++i)
        out[(size_t)(mi * 16 + hf * 8 + i) * S_ + ni * 16 + n] = acc[mi][ni][i];
}

// ---------- kernel 5: row softmax, fp32 scores -> bf16 probabilities ----------

__global__ __launch_bounds__(256) void k_softmax(const float* __restrict__ Sc,
                                                 __bf16* __restrict__ Pb) {
  int row = blockIdx.x;                      // 0 .. B*S-1
  const float* in = Sc + (size_t)row * S_;
  __bf16* out = Pb + (size_t)row * S_;
  int t = threadIdx.x;

  float v[8];
  float mx = -3.0e38f;
#pragma unroll
  for (int i = 0; i < 8; ++i) { v[i] = in[t + i * 256]; mx = fmaxf(mx, v[i]); }
#pragma unroll
  for (int o = 16; o > 0; o >>= 1) mx = fmaxf(mx, __shfl_xor(mx, o, 32));

  __shared__ float redm[8];
  __shared__ float reds[8];
  int wv = t >> 5, ln = t & 31;
  if (ln == 0) redm[wv] = mx;
  __syncthreads();
  mx = redm[0];
#pragma unroll
  for (int i = 1; i < 8; ++i) mx = fmaxf(mx, redm[i]);

  float s = 0.0f;
#pragma unroll
  for (int i = 0; i < 8; ++i) { v[i] = __expf(v[i] - mx); s += v[i]; }
#pragma unroll
  for (int o = 16; o > 0; o >>= 1) s += __shfl_xor(s, o, 32);
  if (ln == 0) reds[wv] = s;
  __syncthreads();
  s = 0.0f;
#pragma unroll
  for (int i = 0; i < 8; ++i) s += reds[i];

  float inv = 1.0f / s;
#pragma unroll
  for (int i = 0; i < 8; ++i) out[t + i * 256] = f2bf(v[i] * inv);
}

// ---------- kernel 6: O = P V  (B-operand from VT), 4x4 blocked (64x64/wave) ----------

__global__ __launch_bounds__(256) void k_pv(const __bf16* __restrict__ Pb,
                                            const __bf16* __restrict__ VT,
                                            float* __restrict__ Out) {
  const int QT = S_ / 64;          // 32
  const int NT = D_ / 64;          // 16
  int wave = blockIdx.x * (blockDim.x >> 5) + (threadIdx.x >> 5);
  int lane = threadIdx.x & 31;
  int b  = wave / (QT * NT);
  int r  = wave % (QT * NT);
  int qt = r / NT, dt = r % NT;

  const __bf16* A0 = Pb + ((size_t)b * S_ + qt * 64) * S_;
  const __bf16* B0 = VT + (size_t)b * D_ * S_ + (size_t)(dt * 64) * S_;

  v8f acc[4][4] = {};
  gemm_tile<4, 4>(A0, S_, B0, S_, S_, lane, acc);

  int n = lane & 15, hf = lane >> 4;
  float* out = Out + ((size_t)b * S_ + qt * 64) * D_ + dt * 64;
#pragma unroll
  for (int mi = 0; mi < 4; ++mi)
#pragma unroll
    for (int ni = 0; ni < 4; ++ni)
#pragma unroll
      for (int i = 0; i < 8; ++i)
        out[(size_t)(mi * 16 + hf * 8 + i) * D_ + ni * 16 + n] = acc[mi][ni][i];
}

// ---------- launcher ----------

extern "C" void kernel_launch(void* const* d_in, const int* in_sizes, int n_in,
                              void* d_out, int out_size, void* d_ws, size_t ws_size,
                              hipStream_t stream) {
  const float* x  = (const float*)d_in[0];
  const float* Wq = (const float*)d_in[1];
  const float* bq = (const float*)d_in[2];
  const float* Wk = (const float*)d_in[3];
  const float* bk = (const float*)d_in[4];
  const float* Wv = (const float*)d_in[5];
  const float* bv = (const float*)d_in[6];
  float* out = (float*)d_out;

  const size_t nBSD = (size_t)B_ * S_ * D_;      // 8.39M elements
  const size_t nBSS = (size_t)B_ * S_ * S_;      // 16.8M elements

  char* p = (char*)d_ws;
  __bf16* xb = (__bf16*)p;  p += nBSD * 2;
  __bf16* Wt = (__bf16*)p;  p += (size_t)3 * D_ * D_ * 2;
  __bf16* Qb = (__bf16*)p;  p += nBSD * 2;
  __bf16* Kb = (__bf16*)p;  p += nBSD * 2;
  __bf16* VT = (__bf16*)p;  p += nBSD * 2;
  float*  Sc = (float*)p;   p += nBSS * 4;
  __bf16* Pb = (__bf16*)p;

  // 1) x -> bf16
  k_convert_x<<<(unsigned)(nBSD / 4 / 256), 256, 0, stream>>>(x, xb);
  // 2) W -> W^T bf16
  k_transpose_w<<<3 * (D_ / 16) * (D_ / 16), 256, 0, stream>>>(Wq, Wk, Wv, Wt);
  // 3) QKV projections (WMMA, 2x2 blocked)
  {
    int waves = 3 * ((B_ * S_) / 32) * (D_ / 32);    // 24576
    k_qkv<<<waves / 8, 256, 0, stream>>>(xb, Wt, bq, bk, bv, Qb, Kb, VT);
  }
  // 4) scores (WMMA, 4x4 blocked)
  {
    int waves = B_ * (S_ / 64) * (S_ / 64);          // 4096
    k_scores<<<waves / 8, 256, 0, stream>>>(Qb, Kb, Sc);
  }
  // 5) softmax
  k_softmax<<<B_ * S_, 256, 0, stream>>>(Sc, Pb);
  // 6) P @ V (WMMA, 4x4 blocked)
  {
    int waves = B_ * (S_ / 64) * (D_ / 64);          // 2048
    k_pv<<<waves / 8, 256, 0, stream>>>(Pb, VT, out);
  }
}
